// MaskedSelfAttention_7181185319383
// MI455X (gfx1250) — compile-verified, matrix operand order checked by reading, not run on hardware
//
#include <hip/hip_runtime.h>
#include <hip/hip_bf16.h>
#include <math.h>

typedef __attribute__((ext_vector_type(16))) _Float16 v16h;
typedef __attribute__((ext_vector_type(8)))  _Float16 v8h;
typedef __attribute__((ext_vector_type(8)))  float    v8f;
typedef __attribute__((ext_vector_type(4)))  float    v4f;

#define D_IN   768
#define D_OUT  16
#define SEQ    2048
#define BATCH  8
#define NCHUNK (D_IN / 32)   // 24 K-chunks of 32

// Wave-local LDS ordering: DS ops are in-order per wave (DScnt); this fence
// stops compiler reordering and waits all outstanding DS ops.
#define WAVE_LDS_FENCE() asm volatile("s_wait_dscnt 0x0" ::: "memory")

// ---------------------------------------------------------------------------
// Prep: swizzle Wq/Wk/Wv (f32 [768 x 16] row-major) into f16 WMMA-B register
// layout: sw[(chunk*32 + lane)*16 + j] = W[(kc + kk(j,half))*16 + n],
// n = lane&15, half = lane>>4,  kk = (j/8)*16 + half*8 + (j%8).
// ---------------------------------------------------------------------------
__global__ __launch_bounds__(32)
void wswizzle_kernel(const float* __restrict__ Wq, const float* __restrict__ Wk,
                     const float* __restrict__ Wv,
                     _Float16* __restrict__ swq, _Float16* __restrict__ swk,
                     _Float16* __restrict__ swv)
{
    const int lane = threadIdx.x;
    const int n    = lane & 15;
    const int half = lane >> 4;
    const int kc   = blockIdx.x * 32;
    const int obase = (blockIdx.x * 32 + lane) * 16;
#pragma unroll
    for (int j = 0; j < 16; ++j) {
        const int kk  = ((j >> 3) << 4) + half * 8 + (j & 7);
        const int idx = (kc + kk) * D_OUT + n;
        swq[obase + j] = (_Float16)Wq[idx];
        swk[obase + j] = (_Float16)Wk[idx];
        swv[obase + j] = (_Float16)Wv[idx];
    }
}

// ---------------------------------------------------------------------------
// Pass 1: QKV projection.  4 independent waves per block, one 16-row tile per
// wave.  A: x via b128 loads + packed cvt.  B: pre-swizzled f16, one 32B load.
// Outputs (f16, pass-2-friendly):
//   Qh row-major pre-scaled, Kh row-major (per-wave LDS transpose),
//   Vt transposed [b][d][s] (D-layout maps directly, b128 stores).
// ---------------------------------------------------------------------------
__global__ __launch_bounds__(128)
void qkv_proj_kernel(const float* __restrict__ x,
                     const _Float16* __restrict__ swq,
                     const _Float16* __restrict__ swk,
                     const _Float16* __restrict__ swv,
                     const float* __restrict__ bq, const float* __restrict__ bk,
                     const float* __restrict__ bv,
                     _Float16* __restrict__ Qh, _Float16* __restrict__ Kh,
                     _Float16* __restrict__ Vt)
{
    __shared__ float tbuf[4][16 * 16];

    const int tid  = threadIdx.x;
    const int wid  = tid >> 5;
    const int lane = tid & 31;
    const int m    = lane & 15;
    const int half = lane >> 4;
    const int tile = blockIdx.x * 4 + wid;       // 16-row tile index
    const int bb   = tile >> 7;                  // 128 tiles per batch
    const int s0   = (tile & 127) * 16;
    const float* xrow = x + ((long)tile * 16 + m) * D_IN;

    const v16h* Bq = (const v16h*)swq;
    const v16h* Bk = (const v16h*)swk;
    const v16h* Bv = (const v16h*)swv;

    v8f accq = {}, acck = {}, accv = {};

    for (int c = 0; c < NCHUNK; ++c) {
        const v4f* xv = (const v4f*)(xrow + c * 32);
        const v4f a0 = xv[half * 2];
        const v4f a1 = xv[half * 2 + 1];
        const v4f a2 = xv[4 + half * 2];
        const v4f a3 = xv[4 + half * 2 + 1];
        v16h a;
#pragma unroll
        for (int j = 0; j < 4; ++j) {
            a[j]      = (_Float16)a0[j];
            a[j + 4]  = (_Float16)a1[j];
            a[j + 8]  = (_Float16)a2[j];
            a[j + 12] = (_Float16)a3[j];
        }
        const v16h bq_ = Bq[c * 32 + lane];
        const v16h bk_ = Bk[c * 32 + lane];
        const v16h bv_ = Bv[c * 32 + lane];

        accq = __builtin_amdgcn_wmma_f32_16x16x32_f16(false, a, false, bq_,
                                                      (short)0, accq, false, false);
        acck = __builtin_amdgcn_wmma_f32_16x16x32_f16(false, a, false, bk_,
                                                      (short)0, acck, false, false);
        accv = __builtin_amdgcn_wmma_f32_16x16x32_f16(false, a, false, bv_,
                                                      (short)0, accv, false, false);
    }

    // ---- V: D-layout column m == Vt row m, contiguous keys -> direct b128
    {
        v8h vv;
#pragma unroll
        for (int r = 0; r < 8; ++r) vv[r] = (_Float16)(accv[r] + bv[m]);
        *(v8h*)&Vt[((long)bb * 16 + m) * SEQ + s0 + half * 8] = vv;
    }
    // ---- Q: per-wave LDS transpose -> row-major f16, pre-scaled by 1/sqrt(16)
#pragma unroll
    for (int r = 0; r < 8; ++r) tbuf[wid][(r + half * 8) * 16 + m] = accq[r] + bq[m];
    WAVE_LDS_FENCE();
    {
        v8h qv;
#pragma unroll
        for (int t = 0; t < 8; ++t)
            qv[t] = (_Float16)(tbuf[wid][m * 16 + half * 8 + t] * 0.25f);
        *(v8h*)&Qh[((long)bb * SEQ + s0 + m) * 16 + half * 8] = qv;
    }
    WAVE_LDS_FENCE();
    // ---- K: per-wave LDS transpose -> row-major f16
#pragma unroll
    for (int r = 0; r < 8; ++r) tbuf[wid][(r + half * 8) * 16 + m] = acck[r] + bk[m];
    WAVE_LDS_FENCE();
    {
        v8h kv;
#pragma unroll
        for (int t = 0; t < 8; ++t) kv[t] = (_Float16)tbuf[wid][m * 16 + half * 8 + t];
        *(v8h*)&Kh[((long)bb * SEQ + s0 + m) * 16 + half * 8] = kv;
    }
}

// ---------------------------------------------------------------------------
// Pass 2: causal flash attention with intra-block split-K.
// Block = 128 threads = 4 waves sharing one (batch, 16-query tile).
// Wave w processes key tiles j ≡ w (mod 4) with private online-softmax state,
// then the block merges the 4 partials with a log-sum-exp combine in LDS.
// Per key tile: one b128 K load, one b128 Vt load, two WMMAs, shfl softmax.
// ---------------------------------------------------------------------------
__global__ __launch_bounds__(128)
void flash_attn_kernel(const _Float16* __restrict__ Qh,
                       const _Float16* __restrict__ Kh,
                       const _Float16* __restrict__ Vt,
                       float* __restrict__ Out)
{
    __shared__ _Float16 pLds[4][16 * 16];
    __shared__ float    zbuf[4][16 * 16];
    __shared__ float    mbuf[4][16];
    __shared__ float    lbuf[4][16];

    const int tid   = threadIdx.x;
    const int wid   = tid >> 5;
    const int lane  = tid & 31;
    const int m     = lane & 15;
    const int half  = lane >> 4;
    const int qtile = blockIdx.x;                 // 0..127
    const int b     = blockIdx.y;                 // 0..7
    const int q0    = qtile * 16;

    // q tile as A operand (already scaled, f16): one b128 load
    v16h qa;
    {
        const v8h qlo = *(const v8h*)&Qh[((long)b * SEQ + q0 + m) * 16 + half * 8];
#pragma unroll
        for (int t = 0; t < 8; ++t) qa[t] = qlo[t];
#pragma unroll
        for (int t = 8; t < 16; ++t) qa[t] = (_Float16)0.0f;
    }

    v8f  zacc = {};
    float mrow[8], lrow[8];
#pragma unroll
    for (int r = 0; r < 8; ++r) { mrow[r] = -INFINITY; lrow[r] = 0.0f; }

    for (int j = wid; j <= qtile; j += 4) {
        const int k0 = j * 16;

        // K_j^T as B: lane = key column, one b128 load
        v16h kb;
        {
            const v8h klo = *(const v8h*)&Kh[((long)b * SEQ + k0 + m) * 16 + half * 8];
#pragma unroll
            for (int t = 0; t < 8; ++t) kb[t] = klo[t];
#pragma unroll
            for (int t = 8; t < 16; ++t) kb[t] = (_Float16)0.0f;
        }
        v8f c0 = {};
        v8f s  = __builtin_amdgcn_wmma_f32_16x16x32_f16(false, qa, false, kb,
                                                        (short)0, c0, false, false);

        // causal mask on diagonal tile (wave-uniform branch: EXEC stays full)
        if (j == qtile) {
#pragma unroll
            for (int r = 0; r < 8; ++r) {
                const int M = r + half * 8;
                if (m > M) s[r] = -INFINITY;
            }
        }

        // online softmax per row (row spans the 16 lanes of this half-wave)
        float pr[8];
#pragma unroll
        for (int r = 0; r < 8; ++r) {
            float v = s[r];
#pragma unroll
            for (int off = 1; off < 16; off <<= 1)
                v = fmaxf(v, __shfl_xor(v, off, 16));
            const float mnew = fmaxf(mrow[r], v);

            float p = (s[r] == -INFINITY) ? 0.0f : __expf(s[r] - mnew);
            float sum = p;
#pragma unroll
            for (int off = 1; off < 16; off <<= 1)
                sum += __shfl_xor(sum, off, 16);

            const float alpha = (mrow[r] == -INFINITY) ? 0.0f : __expf(mrow[r] - mnew);
            lrow[r] = lrow[r] * alpha + sum;
            zacc[r] *= alpha;
            mrow[r] = mnew;
            pr[r]   = p;
        }

        // relay P (D-layout) -> A-layout through per-wave f16 LDS region.
        // Waves run independent trip counts: wave-local DS ordering only.
#pragma unroll
        for (int r = 0; r < 8; ++r)
            pLds[wid][(r + half * 8) * 16 + m] = (_Float16)pr[r];
        WAVE_LDS_FENCE();

        v16h pa;
        {
            const v8h plo = *(const v8h*)&pLds[wid][m * 16 + half * 8];
#pragma unroll
            for (int t = 0; t < 8; ++t) pa[t] = plo[t];
#pragma unroll
            for (int t = 8; t < 16; ++t) pa[t] = (_Float16)0.0f;
        }
        WAVE_LDS_FENCE();

        // V_j as B: Vt row m (= d column), contiguous keys, one b128 load
        v16h vb;
        {
            const v8h vlo = *(const v8h*)&Vt[((long)b * 16 + m) * SEQ + k0 + half * 8];
#pragma unroll
            for (int t = 0; t < 8; ++t) vb[t] = vlo[t];
#pragma unroll
            for (int t = 8; t < 16; ++t) vb[t] = (_Float16)0.0f;
        }

        zacc = __builtin_amdgcn_wmma_f32_16x16x32_f16(false, pa, false, vb,
                                                      (short)0, zacc, false, false);
    }

    // publish per-wave partials
#pragma unroll
    for (int r = 0; r < 8; ++r)
        zbuf[wid][(r + half * 8) * 16 + m] = zacc[r];
    if (m == 0) {
#pragma unroll
        for (int r = 0; r < 8; ++r) {
            mbuf[wid][r + half * 8] = mrow[r];
            lbuf[wid][r + half * 8] = lrow[r];
        }
    }
    __syncthreads();   // uniform: every wave reaches exactly once

    // log-sum-exp merge of the 4 split-K partials; 2 output elems per thread
#pragma unroll
    for (int e = tid; e < 256; e += 128) {
        const int row = e >> 4;
        const int col = e & 15;
        const float ms = fmaxf(fmaxf(mbuf[0][row], mbuf[1][row]),
                               fmaxf(mbuf[2][row], mbuf[3][row]));
        float zsum = 0.0f, lsum = 0.0f;
#pragma unroll
        for (int w = 0; w < 4; ++w) {
            const float mw = mbuf[w][row];
            const float aw = (mw == -INFINITY) ? 0.0f : __expf(mw - ms);
            lsum += aw * lbuf[w][row];
            zsum += aw * zbuf[w][row * 16 + col];
        }
        Out[((long)b * SEQ + q0 + row) * D_OUT + col] = zsum / lsum;
    }
}

extern "C" void kernel_launch(void* const* d_in, const int* in_sizes, int n_in,
                              void* d_out, int out_size, void* d_ws, size_t ws_size,
                              hipStream_t stream)
{
    const float* x  = (const float*)d_in[0];
    const float* Wq = (const float*)d_in[1];
    const float* bq = (const float*)d_in[2];
    const float* Wk = (const float*)d_in[3];
    const float* bk = (const float*)d_in[4];
    const float* Wv = (const float*)d_in[5];
    const float* bv = (const float*)d_in[6];
    float* out = (float*)d_out;

    // scratch carve-up (f16 elements)
    const long wsw_elems = (long)NCHUNK * 32 * 16;        // 12288 per matrix
    const long qkv_elems = (long)BATCH * SEQ * D_OUT;     // 262144 per tensor
    _Float16* swq = (_Float16*)d_ws;
    _Float16* swk = swq + wsw_elems;
    _Float16* swv = swk + wsw_elems;
    _Float16* Qh  = swv + wsw_elems;
    _Float16* Kh  = Qh + qkv_elems;
    _Float16* Vt  = Kh + qkv_elems;

    // Prep: 24 chunks, one wave each
    wswizzle_kernel<<<dim3(NCHUNK), dim3(32), 0, stream>>>(Wq, Wk, Wv, swq, swk, swv);

    // Pass 1: 1024 row tiles, 4 waves (4 tiles) per block
    qkv_proj_kernel<<<dim3((BATCH * SEQ) / 16 / 4), dim3(128), 0, stream>>>(
        x, swq, swk, swv, bq, bk, bv, Qh, Kh, Vt);

    // Pass 2: 128 query tiles x 8 batches, 4 split-K waves per block
    flash_attn_kernel<<<dim3(SEQ / 16, BATCH), dim3(128), 0, stream>>>(Qh, Kh, Vt, out);
}
